// RNN_70970039599178
// MI455X (gfx1250) — matrix operand order repaired and checked by reading, not compiled, not run on hardware
//
#include <hip/hip_runtime.h>
#include <math.h>

typedef _Float16 half_t;
typedef __attribute__((ext_vector_type(16))) _Float16 v16h;
typedef __attribute__((ext_vector_type(8)))  _Float16 v8h;
typedef __attribute__((ext_vector_type(8)))  float    v8f;
typedef __attribute__((ext_vector_type(4)))  unsigned int v4u;
typedef __attribute__((ext_vector_type(4)))  int      v4i;
typedef __attribute__((ext_vector_type(8)))  int      v8i;

#define S_LEN 512
#define BATCH 64
#define EDIM  300
#define EPAD  320           // E padded to multiple of 32 (WMMA K-step), zeros
#define HDIM  256
#define CDIM  5

// ---- workspace layout (bytes) ----
// xW stored as f16 in WMMA C-fragment order: [blockM*16+tn][lane][8 halves]
#define XWF_BYTES  ((size_t)S_LEN * BATCH * HDIM * 2)   // 16,777,216
#define WIH_BYTES  ((size_t)HDIM * EPAD * 2)            //    163,840

#define A_STRIDE   336      // halves; 672B = 42*16 -> 16B aligned chunks
#define WHH_STRIDE 264      // halves; 528B, keeps B-frag rows on distinct banks
#define H_STRIDE   264

#define XSLAB_HALVES (BATCH * HDIM)            // 16384 halves = 32 KB per step
#define XSLAB_DWORDS (BATCH * HDIM * 2 / 4)    // 8192 dwords

#if defined(__has_builtin)
#if __has_builtin(__builtin_amdgcn_tensor_load_to_lds) && __has_builtin(__builtin_amdgcn_s_wait_tensorcnt)
#define HAVE_TDM 1
#endif
#endif
#ifndef HAVE_TDM
#define HAVE_TDM 0
#endif

#if HAVE_TDM
// Issue one TDM DMA: copy `ndw` dwords from global `gsrc` into LDS offset `lds_off`.
// D# per CDNA5 ISA: group0 = {count|flags, lds_addr,
// global_addr[31:0], global_addr[56:32]|type=2}; group1 packs data_size=4B,
// tensor_dim0=tile_dim0=ndw, tensor_dim1=tile_dim1=1, dim0_stride=ndw.
__device__ inline void tdm_copy_to_lds(const void* gsrc, unsigned int lds_off, unsigned int ndw) {
    unsigned long long ga = (unsigned long long)(uintptr_t)gsrc;
    v4u g0 = { 1u,                                   // count=1 (valid user D#)
               lds_off,                              // lds_addr (bytes)
               (unsigned int)ga,                     // global_addr[31:0]
               (unsigned int)((ga >> 32) & 0x01FFFFFFu) | (2u << 30) };  // addr[56:32]|type=2
    v8i g1 = { (int)(2u << 16),                      // data_size=2 (4-byte elems)
               (int)((ndw & 0xFFFFu) << 16),         // tensor_dim0[15:0] in bits[63:48]
               (int)(((ndw >> 16) & 0xFFFFu) | (1u << 16)), // tensor_dim0[31:16] | tensor_dim1 lo=1
               (int)((ndw & 0xFFFFu) << 16),         // tensor_dim1 hi=0 | tile_dim0=ndw
               1,                                    // tile_dim1=1 | tile_dim2=0
               (int)ndw,                             // tensor_dim0_stride[31:0]
               0, 0 };
    v4i gz = { 0, 0, 0, 0 };
#if __clang_major__ >= 23
    v8i gz8 = { 0, 0, 0, 0, 0, 0, 0, 0 };
    __builtin_amdgcn_tensor_load_to_lds(g0, g1, gz, gz, gz8, 0);
#else
    __builtin_amdgcn_tensor_load_to_lds(g0, g1, gz, gz, 0);
#endif
}
#endif

// ---------------- Phase 0: convert W_ih (H x E f32) -> f16 (H x EPAD, zero pad)
__global__ void prep_wih(const float* __restrict__ Wih, half_t* __restrict__ out) {
    int i = blockIdx.x * blockDim.x + threadIdx.x;
    if (i >= HDIM * EPAD) return;
    int h = i / EPAD, k = i - h * EPAD;
    float v = (k < EDIM) ? Wih[h * EDIM + k] : 0.f;
    out[i] = (half_t)v;
}

// ---------------- Phase 1: xW = gather(emb, inputs) @ W_ih^T + b_ih  (WMMA f16->f32)
// grid = 2048 blocks (one 16-row M tile each), 256 threads = 8 waves, 2 N-tiles/wave.
// Output written as f16 C-fragments so phase 2 slabs are contiguous.
__global__ __launch_bounds__(256)
void gemm1(const int* __restrict__ inputs, const float* __restrict__ emb,
           const half_t* __restrict__ wih, const float* __restrict__ b_ih,
           half_t* __restrict__ xwf) {
    __shared__ half_t As[16 * A_STRIDE];
    const int tid  = threadIdx.x;
    const int wave = tid >> 5, lane = tid & 31;
    const int l = lane & 15, hi = lane >> 4;
    const int blockM = blockIdx.x;                       // 0..2047

    // cooperative gather + f32->f16 convert of 16 rows of X into LDS (zero-padded K)
    for (int i = tid; i < 16 * EPAD; i += 256) {
        int r = i / EPAD, k = i - r * EPAD;
        int vidx = inputs[blockM * 16 + r];
        float v = (k < EDIM) ? emb[(size_t)vidx * EDIM + k] : 0.f;
        As[r * A_STRIDE + k] = (half_t)v;
    }
    __syncthreads();

    const int tn0 = wave * 2, tn1 = wave * 2 + 1;        // N tiles 0..15
    const int kh  = hi * 16;
    const half_t* brow0 = wih + (size_t)(tn0 * 16 + l) * EPAD + kh;
    const half_t* brow1 = wih + (size_t)(tn1 * 16 + l) * EPAD + kh;
    const half_t* arow  = As + l * A_STRIDE + hi * 16;

    v8f c0 = {}; v8f c1 = {};
    #pragma unroll
    for (int kb = 0; kb < EPAD; kb += 32) {
        v8h alo = *(const v8h*)(arow + kb);
        v8h ahi = *(const v8h*)(arow + kb + 8);
        v16h a;
        #pragma unroll
        for (int i = 0; i < 8; ++i) { a[i] = alo[i]; a[i + 8] = ahi[i]; }
        v16h b0 = *(const v16h*)(brow0 + kb);
        v16h b1 = *(const v16h*)(brow1 + kb);
        c0 = __builtin_amdgcn_wmma_f32_16x16x32_f16(false, a, false, b0, (short)0, c0, false, false);
        c1 = __builtin_amdgcn_wmma_f32_16x16x32_f16(false, a, false, b1, (short)0, c1, false, false);
    }
    const float bias0 = b_ih[tn0 * 16 + l];
    const float bias1 = b_ih[tn1 * 16 + l];
    v8h p0, p1;
    #pragma unroll
    for (int v = 0; v < 8; ++v) {
        p0[v] = (half_t)(c0[v] + bias0);
        p1[v] = (half_t)(c1[v] + bias1);
    }
    *(v8h*)(xwf + (((size_t)blockM * 16 + tn0) * 32 + lane) * 8) = p0;
    *(v8h*)(xwf + (((size_t)blockM * 16 + tn1) * 32 + lane) * 8) = p1;
}

// ---------------- Phase 2: sequential RNN scan, single persistent workgroup.
// W_hh f16 resident in LDS; h double-buffered in LDS; per-step xW slab streamed
// into double-buffered LDS via the Tensor Data Mover; hsum kept in registers.
__global__ __launch_bounds__(1024)
void rnn_scan(const float* __restrict__ Whh, const float* __restrict__ b_hh,
              const half_t* __restrict__ xwf, float* __restrict__ hsum_out) {
    extern __shared__ half_t lds[];
    half_t* whh   = lds;                                 // [HDIM][WHH_STRIDE]
    half_t* hbuf0 = whh + HDIM * WHH_STRIDE;             // [BATCH][H_STRIDE]
    half_t* hbuf1 = hbuf0 + BATCH * H_STRIDE;
    half_t* xbuf0 = hbuf1 + BATCH * H_STRIDE;            // XSLAB_HALVES each
    half_t* xbuf1 = xbuf0 + XSLAB_HALVES;

    const int tid  = threadIdx.x;
    const int wave = tid >> 5, lane = tid & 31;
    const int l = lane & 15, hi = lane >> 4;

#if HAVE_TDM
    const unsigned int xoff0 = (unsigned int)(uintptr_t)xbuf0;
    const unsigned int xoff1 = (unsigned int)(uintptr_t)xbuf1;
    if (wave == 0)  // kick off slab for t=0 while we stage weights
        tdm_copy_to_lds(xwf, xoff0, XSLAB_DWORDS);
#endif

    // stage W_hh (row n = W_hh[n][:], which is column n of B = W_hh^T)
    for (int i = tid; i < HDIM * HDIM; i += 1024) {
        int n = i >> 8, k = i & 255;
        whh[n * WHH_STRIDE + k] = (half_t)Whh[i];
    }
    for (int i = tid; i < BATCH * H_STRIDE; i += 1024) hbuf0[i] = (half_t)0.f;
#if HAVE_TDM
    if (wave == 0) __builtin_amdgcn_s_wait_tensorcnt(0);
#endif
    __syncthreads();

    const int t0 = wave * 2, t1 = wave * 2 + 1;          // tile ids 0..63
    const int tm = t0 >> 4;                              // 0..3 (shared by pair)
    const int tn0 = t0 & 15, tn1 = t1 & 15;
    const int n0 = tn0 * 16 + l, n1 = tn1 * 16 + l;
    const float bhh0 = b_hh[n0], bhh1 = b_hh[n1];
    const half_t* b0p = whh + n0 * WHH_STRIDE + hi * 16;
    const half_t* b1p = whh + n1 * WHH_STRIDE + hi * 16;
    const int f0 = ((tm * 16 + tn0) * 32 + lane) * 8;    // frag offsets within slab
    const int f1 = ((tm * 16 + tn1) * 32 + lane) * 8;

    v8f hs0 = {}; v8f hs1 = {};
    half_t* hcur = hbuf0; half_t* hnext = hbuf1;
    half_t* xcur = xbuf0; half_t* xnxt = xbuf1;

    for (int t = 0; t < S_LEN; ++t) {
#if HAVE_TDM
        if (wave == 0 && t + 1 < S_LEN)   // async DMA of next step's slab
            tdm_copy_to_lds(xwf + (size_t)(t + 1) * XSLAB_HALVES,
                            (unsigned int)(uintptr_t)xnxt, XSLAB_DWORDS);
        v8h x0 = *(const v8h*)(xcur + f0);
        v8h x1 = *(const v8h*)(xcur + f1);
#else
        v8h x0 = *(const v8h*)(xwf + (size_t)t * XSLAB_HALVES + f0);
        v8h x1 = *(const v8h*)(xwf + (size_t)t * XSLAB_HALVES + f1);
#endif
        v8f c0, c1;
        #pragma unroll
        for (int v = 0; v < 8; ++v) { c0[v] = (float)x0[v]; c1[v] = (float)x1[v]; }

        const half_t* arow = hcur + (tm * 16 + l) * H_STRIDE + hi * 16;
        #pragma unroll
        for (int kb = 0; kb < HDIM; kb += 32) {
            v8h alo = *(const v8h*)(arow + kb);
            v8h ahi = *(const v8h*)(arow + kb + 8);
            v16h a;
            #pragma unroll
            for (int i = 0; i < 8; ++i) { a[i] = alo[i]; a[i + 8] = ahi[i]; }
            v16h b0 = *(const v16h*)(b0p + kb);
            v16h b1 = *(const v16h*)(b1p + kb);
            c0 = __builtin_amdgcn_wmma_f32_16x16x32_f16(false, a, false, b0, (short)0, c0, false, false);
            c1 = __builtin_amdgcn_wmma_f32_16x16x32_f16(false, a, false, b1, (short)0, c1, false, false);
        }
        half_t* o0 = hnext + (tm * 16 + 8 * hi) * H_STRIDE + n0;
        half_t* o1 = hnext + (tm * 16 + 8 * hi) * H_STRIDE + n1;
        #pragma unroll
        for (int v = 0; v < 8; ++v) {
            float th0 = tanhf(c0[v] + bhh0);
            float th1 = tanhf(c1[v] + bhh1);
            hs0[v] += th0;  hs1[v] += th1;
            o0[v * H_STRIDE] = (half_t)th0;
            o1[v * H_STRIDE] = (half_t)th1;
        }
#if HAVE_TDM
        if (wave == 0) __builtin_amdgcn_s_wait_tensorcnt(0);
#endif
        __syncthreads();
        half_t* tmp = hcur; hcur = hnext; hnext = tmp;
        half_t* xt = xcur; xcur = xnxt; xnxt = xt;
    }

    float* s0 = hsum_out + (size_t)(tm * 16 + 8 * hi) * HDIM + n0;
    float* s1 = hsum_out + (size_t)(tm * 16 + 8 * hi) * HDIM + n1;
    #pragma unroll
    for (int v = 0; v < 8; ++v) { s0[(size_t)v * HDIM] = hs0[v]; s1[(size_t)v * HDIM] = hs1[v]; }
}

// ---------------- Phase 3: logits = hsum @ W_out^T + S*b_out ; log_softmax
__global__ void head_kernel(const float* __restrict__ hsum, const float* __restrict__ Wout,
                            const float* __restrict__ bout, float* __restrict__ out) {
    int b = blockIdx.x * blockDim.x + threadIdx.x;
    if (b >= BATCH) return;
    float logits[CDIM];
    #pragma unroll
    for (int c = 0; c < CDIM; ++c) {
        float acc = (float)S_LEN * bout[c];
        for (int h = 0; h < HDIM; ++h) acc += hsum[b * HDIM + h] * Wout[c * HDIM + h];
        logits[c] = acc;
    }
    float m = logits[0];
    #pragma unroll
    for (int c = 1; c < CDIM; ++c) m = fmaxf(m, logits[c]);
    float s = 0.f;
    #pragma unroll
    for (int c = 0; c < CDIM; ++c) s += expf(logits[c] - m);
    float lse = m + logf(s);
    #pragma unroll
    for (int c = 0; c < CDIM; ++c) out[b * CDIM + c] = logits[c] - lse;
}

extern "C" void kernel_launch(void* const* d_in, const int* in_sizes, int n_in,
                              void* d_out, int out_size, void* d_ws, size_t ws_size,
                              hipStream_t stream) {
    (void)in_sizes; (void)n_in; (void)out_size; (void)ws_size;
    const int*   inputs = (const int*)  d_in[0];
    const float* emb    = (const float*)d_in[1];
    const float* W_ih   = (const float*)d_in[2];
    const float* W_hh   = (const float*)d_in[3];
    const float* b_ih   = (const float*)d_in[4];
    const float* b_hh   = (const float*)d_in[5];
    const float* W_out  = (const float*)d_in[6];
    const float* b_out  = (const float*)d_in[7];
    float* out = (float*)d_out;

    char* ws = (char*)d_ws;
    half_t* xwf  = (half_t*)(ws);
    half_t* wih  = (half_t*)(ws + XWF_BYTES);
    float*  hsum = (float*) (ws + XWF_BYTES + WIH_BYTES);

    prep_wih<<<(HDIM * EPAD + 255) / 256, 256, 0, stream>>>(W_ih, wih);
    gemm1<<<(S_LEN * BATCH) / 16, 256, 0, stream>>>(inputs, emb, wih, b_ih, xwf);

    size_t ldsB = (size_t)(HDIM * WHH_STRIDE + 2 * BATCH * H_STRIDE + 2 * XSLAB_HALVES)
                  * sizeof(half_t);   // 268,288 bytes < 320 KB
    hipFuncSetAttribute(reinterpret_cast<const void*>(rnn_scan),
                        hipFuncAttributeMaxDynamicSharedMemorySize, (int)ldsB);
    rnn_scan<<<1, 1024, ldsB, stream>>>(W_hh, b_hh, xwf, hsum);

    head_kernel<<<1, 64, 0, stream>>>(hsum, W_out, b_out, out);
}
